// BiLSTM_CRF_37778532336447
// MI455X (gfx1250) — matrix operand order repaired and checked by reading, not compiled, not run on hardware
//
#include <hip/hip_runtime.h>
#include <hip/hip_bf16.h>
#include <math.h>

// ---------------------------------------------------------------------------
// BiLSTM-CRF forward for MI455X (gfx1250, wave32, WMMA).
//  - gates GEMM + emission GEMM: v_wmma_f32_16x16x32_f16
//  - LSTM recurrence: LDS-resident weight slices (loaded once), agent-scope
//    grid barrier per step (sequential dependence; WMMA not applicable at N=1)
// ---------------------------------------------------------------------------

typedef __attribute__((ext_vector_type(16))) _Float16 v16h;
typedef __attribute__((ext_vector_type(8)))  _Float16 v8h;
typedef __attribute__((ext_vector_type(8)))  float    v8f;

#define S_LEN   4096
#define EDIM    256
#define HDIM    512
#define G4      2048      // 4*HDIM
#define NTAGS   16
#define START_TAG 14
#define STOP_TAG  15
#define NEGINF  (-10000.0f)
#define LSTM_BLOCKS 64
#define UNITS_PER_BLOCK 8          // hidden units per block
#define ROWS_PER_BLOCK  32         // 4 gates * 8 units
#define WPAD (HDIM + 8)            // padded LDS row stride (halves)

// ---------------------------------------------------------------------------
// WMMA fragment helpers (layouts per CDNA5 ISA §7.12.2)
// ---------------------------------------------------------------------------

// A: 16x32 (MxK) f16 from row-major A[lda]. Lanes 0-15 -> K {0..7,16..23},
// lanes 16-31 -> K {8..15,24..31}; two global_load_b128 per lane.
__device__ __forceinline__ v16h load_a_frag(const _Float16* __restrict__ A,
                                            int lda, int m0, int k0) {
  const int lane = threadIdx.x & 31;
  const int r = lane & 15, hh = lane >> 4;
  const _Float16* p = A + (size_t)(m0 + r) * lda + k0 + hh * 8;
  v8h lo = *reinterpret_cast<const v8h*>(p);
  v8h hi = *reinterpret_cast<const v8h*>(p + 16);
  return __builtin_shufflevector(lo, hi, 0,1,2,3,4,5,6,7,8,9,10,11,12,13,14,15);
}

// B: 32x16 (KxN), B[k][n] = W[n][k] (W is row-major [N][ldw]).
// Lanes 0-15 hold K {0..7,16..23}, lanes 16-31 hold K {8..15,24..31}; col = lane&15.
__device__ __forceinline__ v16h load_b_frag(const _Float16* __restrict__ W,
                                            int ldw, int n0, int k0) {
  const int lane = threadIdx.x & 31;
  const int c = lane & 15, hh = lane >> 4;
  const _Float16* p = W + (size_t)(n0 + c) * ldw + k0 + hh * 8;
  v8h lo = *reinterpret_cast<const v8h*>(p);
  v8h hi = *reinterpret_cast<const v8h*>(p + 16);
  return __builtin_shufflevector(lo, hi, 0,1,2,3,4,5,6,7,8,9,10,11,12,13,14,15);
}

// C/D: 8 VGPRs f32; vgpr r -> M = r + 8*(lane>=16), N = lane&15.
__device__ __forceinline__ void store_c_bias(float* __restrict__ C, int ldc,
                                             int m0, int n0, const v8f acc,
                                             const float* __restrict__ bias) {
  const int lane = threadIdx.x & 31;
  const int c = lane & 15, hh = lane >> 4;
  const float bn = bias[n0 + c];
#pragma unroll
  for (int r = 0; r < 8; ++r)
    C[(size_t)(m0 + r + hh * 8) * ldc + n0 + c] = acc[r] + bn;
}

// ---------------------------------------------------------------------------
// Prep kernels
// ---------------------------------------------------------------------------

__global__ void embed_gather_f16(const int* __restrict__ sent,
                                 const float* __restrict__ emb,
                                 _Float16* __restrict__ out) {
  int idx = blockIdx.x * blockDim.x + threadIdx.x;
  if (idx >= S_LEN * EDIM) return;
  int s = idx >> 8, k = idx & 255;
  out[idx] = (_Float16)emb[(size_t)sent[s] * EDIM + k];
}

__global__ void cvt_f32_to_f16(const float* __restrict__ in,
                               _Float16* __restrict__ out, int n) {
  int idx = blockIdx.x * blockDim.x + threadIdx.x;
  if (idx < n) out[idx] = (_Float16)in[idx];
}

__global__ void add_vec(const float* __restrict__ a, const float* __restrict__ b,
                        float* __restrict__ o, int n) {
  int idx = blockIdx.x * blockDim.x + threadIdx.x;
  if (idx < n) o[idx] = a[idx] + b[idx];
}

__global__ void init_barrier(unsigned* b) {
  if (threadIdx.x == 0) { b[0] = 0u; b[32] = 0u; }
}

// ---------------------------------------------------------------------------
// GEMM 1: gates_x[S][2048] = embA[S][256] @ w_ih^T + (b_ih+b_hh)
// 8 waves/block, each wave: 1 M-tile x 4 N-tiles, K-loop of 4 WMMAs/step.
// ---------------------------------------------------------------------------

__global__ void __launch_bounds__(256) gemm_gates(const _Float16* __restrict__ A,
                                                  const _Float16* __restrict__ W,
                                                  const float* __restrict__ bias,
                                                  float* __restrict__ C) {
  const int wave = threadIdx.x >> 5;
  const int job = blockIdx.x * 8 + wave;       // 8192 jobs = 256 mt * 32 ng
  const int mt = job >> 5;
  const int ng = job & 31;
  const int m0 = mt * 16;
  const int n0 = ng * 64;

  v8f acc0 = {}, acc1 = {}, acc2 = {}, acc3 = {};
  for (int k0 = 0; k0 < EDIM; k0 += 32) {
    if (k0 + 32 < EDIM) {
      const int lane = threadIdx.x & 31;
      __builtin_prefetch(A + (size_t)(m0 + (lane & 15)) * EDIM + k0 + 32, 0, 3);
      __builtin_prefetch(W + (size_t)(n0 + (lane & 15)) * EDIM + k0 + 32, 0, 3);
    }
    v16h a  = load_a_frag(A, EDIM, m0, k0);
    v16h b0 = load_b_frag(W, EDIM, n0,      k0);
    v16h b1 = load_b_frag(W, EDIM, n0 + 16, k0);
    v16h b2 = load_b_frag(W, EDIM, n0 + 32, k0);
    v16h b3 = load_b_frag(W, EDIM, n0 + 48, k0);
    acc0 = __builtin_amdgcn_wmma_f32_16x16x32_f16(false, a, false, b0, (short)0, acc0, false, false);
    acc1 = __builtin_amdgcn_wmma_f32_16x16x32_f16(false, a, false, b1, (short)0, acc1, false, false);
    acc2 = __builtin_amdgcn_wmma_f32_16x16x32_f16(false, a, false, b2, (short)0, acc2, false, false);
    acc3 = __builtin_amdgcn_wmma_f32_16x16x32_f16(false, a, false, b3, (short)0, acc3, false, false);
  }
  store_c_bias(C, G4, m0, n0,      acc0, bias);
  store_c_bias(C, G4, m0, n0 + 16, acc1, bias);
  store_c_bias(C, G4, m0, n0 + 32, acc2, bias);
  store_c_bias(C, G4, m0, n0 + 48, acc3, bias);
}

// ---------------------------------------------------------------------------
// LSTM recurrence: 4096 sequential steps. Block b owns hidden units
// [b*8, b*8+8) (all 4 gates = 32 rows of w_hh). The 32x512 f16 weight slice
// is loaded into LDS ONCE (weights are step-invariant), so the per-step work
// reads only LDS + the 8KB gates_x row; no L2 weight streaming.
// Thread map: rl = tid&31 (gate = rl>>3, unit = rl&7), ksub = tid>>5 (8 x 64k).
// ---------------------------------------------------------------------------

__global__ void __launch_bounds__(256) lstm_rec(const float* __restrict__ gates,
                                                const float* __restrict__ whh,
                                                const float* __restrict__ h0,
                                                const float* __restrict__ c0,
                                                float* __restrict__ hbuf,
                                                _Float16* __restrict__ hs,
                                                unsigned* __restrict__ bar) {
  __shared__ _Float16 w_sh[ROWS_PER_BLOCK * WPAD];   // ~33 KB, padded rows
  __shared__ float h_sh[HDIM];
  __shared__ float red[256];
  __shared__ float gsum[ROWS_PER_BLOCK];
  __shared__ float c_sh[UNITS_PER_BLOCK];

  const int tid  = threadIdx.x;
  const int rl   = tid & 31;                  // local row: gate*8 + unit
  const int gate = rl >> 3;
  const int u    = rl & 7;
  const int ksub = tid >> 5;                  // 0..7, 64 K each
  const int u0   = blockIdx.x * UNITS_PER_BLOCK;
  const int row  = gate * HDIM + u0 + u;      // torch gate order i,f,g,o
  unsigned* cnt = bar;
  unsigned* gen = bar + 32;

  // One-time weight preload: w_sh[r][k] = (f16) w_hh[gate*512 + u0 + unit][k]
  for (int idx = tid; idx < ROWS_PER_BLOCK * HDIM; idx += 256) {
    int r = idx >> 9;                          // / HDIM
    int k = idx & (HDIM - 1);
    int g = r >> 3, uu = r & 7;
    w_sh[r * WPAD + k] = (_Float16)whh[(size_t)(g * HDIM + u0 + uu) * HDIM + k];
  }
  if (tid < UNITS_PER_BLOCK) c_sh[tid] = c0[u0 + tid];

  const _Float16* wrow = &w_sh[rl * WPAD];
  const int kbase = ksub * 64;

  for (int s = 0; s < S_LEN; ++s) {
    const float* hsrc = (s == 0) ? h0 : (hbuf + (s & 1) * HDIM);
    for (int i = tid; i < HDIM; i += 256) h_sh[i] = hsrc[i];
    __syncthreads();

    // 64 MACs per thread, weights via 16B ds_load_b128
    float acc = 0.f;
#pragma unroll
    for (int kk = 0; kk < 64; kk += 8) {
      v8h wv = *reinterpret_cast<const v8h*>(wrow + kbase + kk);
#pragma unroll
      for (int j = 0; j < 8; ++j)
        acc += (float)wv[j] * h_sh[kbase + kk + j];
    }
    red[tid] = acc;
    __syncthreads();

    if (tid < 32) {
      float tot = gates[(size_t)s * G4 + row];
#pragma unroll
      for (int j = 0; j < 8; ++j) tot += red[tid + 32 * j];
      gsum[tid] = tot;                         // gsum[gate*8 + unit]
    }
    __syncthreads();

    if (tid < UNITS_PER_BLOCK) {
      float ig = 1.f / (1.f + __expf(-gsum[tid]));
      float fg = 1.f / (1.f + __expf(-gsum[8 + tid]));
      float gg = tanhf(gsum[16 + tid]);
      float og = 1.f / (1.f + __expf(-gsum[24 + tid]));
      float c  = fg * c_sh[tid] + ig * gg;
      c_sh[tid] = c;
      float h  = og * tanhf(c);
      hbuf[((s + 1) & 1) * HDIM + u0 + tid] = h;
      hs[(size_t)s * HDIM + u0 + tid] = (_Float16)h;
    }
    __threadfence();
    __syncthreads();
    if (tid == 0) {
      unsigned g = __hip_atomic_load(gen, __ATOMIC_RELAXED, __HIP_MEMORY_SCOPE_AGENT);
      unsigned a = __hip_atomic_fetch_add(cnt, 1u, __ATOMIC_ACQ_REL, __HIP_MEMORY_SCOPE_AGENT);
      if (a == LSTM_BLOCKS - 1) {
        __hip_atomic_store(cnt, 0u, __ATOMIC_RELAXED, __HIP_MEMORY_SCOPE_AGENT);
        __hip_atomic_store(gen, g + 1u, __ATOMIC_RELEASE, __HIP_MEMORY_SCOPE_AGENT);
      } else {
        while (__hip_atomic_load(gen, __ATOMIC_ACQUIRE, __HIP_MEMORY_SCOPE_AGENT) == g)
          __builtin_amdgcn_s_sleep(2);
      }
    }
    __syncthreads();
  }
}

// ---------------------------------------------------------------------------
// GEMM 2: emissions[S][16] = hs[S][512] @ w_out^T + b_out (N = one WMMA tile)
// ---------------------------------------------------------------------------

__global__ void __launch_bounds__(256) gemm_emis(const _Float16* __restrict__ Hs,
                                                 const _Float16* __restrict__ Wout,
                                                 const float* __restrict__ bout,
                                                 float* __restrict__ E) {
  const int wave = threadIdx.x >> 5;
  const int mt = blockIdx.x * 8 + wave;       // 256 m-tiles
  const int m0 = mt * 16;
  v8f acc = {};
  for (int k0 = 0; k0 < HDIM; k0 += 32) {
    if (k0 + 32 < HDIM) {
      const int lane = threadIdx.x & 31;
      __builtin_prefetch(Hs + (size_t)(m0 + (lane & 15)) * HDIM + k0 + 32, 0, 3);
    }
    v16h a = load_a_frag(Hs, HDIM, m0, k0);
    v16h b = load_b_frag(Wout, HDIM, 0, k0);
    acc = __builtin_amdgcn_wmma_f32_16x16x32_f16(false, a, false, b, (short)0, acc, false, false);
  }
  store_c_bias(E, NTAGS, m0, 0, acc, bout);
}

// ---------------------------------------------------------------------------
// Viterbi: one wave; 16 lanes = tags; transitions in LDS; serial backtrace.
// ---------------------------------------------------------------------------

__global__ void __launch_bounds__(32) viterbi(const float* __restrict__ em,
                                              const float* __restrict__ trans,
                                              unsigned char* __restrict__ bp,
                                              float* __restrict__ out) {
  __shared__ float tr[NTAGS * NTAGS];
  __shared__ float fs[NTAGS], fsn[NTAGS];
  const int t = threadIdx.x;
  for (int i = t; i < NTAGS * NTAGS; i += 32) tr[i] = trans[i];
  if (t < NTAGS) fs[t] = (t == START_TAG) ? 0.f : NEGINF;
  __syncthreads();

  for (int s = 0; s < S_LEN; ++s) {
    if (t < NTAGS) {
      float best = -INFINITY; int bb = 0;
#pragma unroll
      for (int p = 0; p < NTAGS; ++p) {
        float v = fs[p] + tr[p * NTAGS + t];
        if (v > best) { best = v; bb = p; }
      }
      fsn[t] = best + em[(size_t)s * NTAGS + t];
      bp[(size_t)s * NTAGS + t] = (unsigned char)bb;
    }
    __syncthreads();
    if (t < NTAGS) fs[t] = fsn[t];
    __syncthreads();
  }

  if (t == 0) {
    float best = -INFINITY; int bb = 0;
    for (int p = 0; p < NTAGS; ++p) {
      float v = fs[p] + tr[p * NTAGS + STOP_TAG];
      if (v > best) { best = v; bb = p; }
    }
    out[0] = best;
    int cur = bb;
    for (int s = S_LEN - 1; s >= 0; --s) {
      out[1 + s] = (float)cur;
      cur = (int)bp[(size_t)s * NTAGS + cur];
    }
  }
}

// ---------------------------------------------------------------------------
// Host orchestration
// ---------------------------------------------------------------------------

extern "C" void kernel_launch(void* const* d_in, const int* in_sizes, int n_in,
                              void* d_out, int out_size, void* d_ws, size_t ws_size,
                              hipStream_t stream) {
  const int*   sent  = (const int*)  d_in[0];
  const float* emb   = (const float*)d_in[1];
  const float* wih   = (const float*)d_in[2];
  const float* whh   = (const float*)d_in[3];
  const float* bih   = (const float*)d_in[4];
  const float* bhh   = (const float*)d_in[5];
  const float* wout  = (const float*)d_in[6];
  const float* bout  = (const float*)d_in[7];
  const float* trans = (const float*)d_in[8];
  const float* h0    = (const float*)d_in[9];
  const float* c0    = (const float*)d_in[10];
  float* out = (float*)d_out;

  char* ws = (char*)d_ws;
  size_t off = 0;
  auto take = [&](size_t bytes) -> void* {
    void* p = ws + off;
    off = (off + bytes + 255) & ~(size_t)255;
    return p;
  };
  _Float16* embA   = (_Float16*)take((size_t)S_LEN * EDIM * 2);   // 2 MB
  _Float16* wih16  = (_Float16*)take((size_t)G4 * EDIM * 2);      // 1 MB
  float*    bvec   = (float*)   take((size_t)G4 * 4);
  float*    gatesx = (float*)   take((size_t)S_LEN * G4 * 4);     // 32 MB
  _Float16* hs16   = (_Float16*)take((size_t)S_LEN * HDIM * 2);   // 4 MB
  _Float16* wout16 = (_Float16*)take((size_t)NTAGS * HDIM * 2);
  float*    emis   = (float*)   take((size_t)S_LEN * NTAGS * 4);
  float*    hbuf   = (float*)   take((size_t)2 * HDIM * 4);
  unsigned char* bpbuf = (unsigned char*)take((size_t)S_LEN * NTAGS);
  unsigned* barm   = (unsigned*)take(256);

  // Prep
  embed_gather_f16<<<(S_LEN * EDIM + 255) / 256, 256, 0, stream>>>(sent, emb, embA);
  cvt_f32_to_f16<<<(G4 * EDIM + 255) / 256, 256, 0, stream>>>(wih, wih16, G4 * EDIM);
  cvt_f32_to_f16<<<(NTAGS * HDIM + 255) / 256, 256, 0, stream>>>(wout, wout16, NTAGS * HDIM);
  add_vec<<<(G4 + 255) / 256, 256, 0, stream>>>(bih, bhh, bvec, G4);
  init_barrier<<<1, 32, 0, stream>>>(barm);

  // Input projection GEMM (WMMA): 8192 wave-jobs / 8 waves per block
  gemm_gates<<<1024, 256, 0, stream>>>(embA, wih16, bvec, gatesx);

  // Sequential LSTM recurrence: LDS-resident weights + grid barrier per step
  lstm_rec<<<LSTM_BLOCKS, 256, 0, stream>>>(gatesx, whh, h0, c0, hbuf, hs16, barm);

  // Emission GEMM (WMMA, N = one tile)
  gemm_emis<<<32, 256, 0, stream>>>(hs16, wout16, bout, emis);

  // Viterbi decode + backtrace
  viterbi<<<1, 32, 0, stream>>>(emis, trans, bpbuf, out);
}
